// VQCodebook_22290880266350
// MI455X (gfx1250) — matrix operand order repaired and checked by reading, not compiled, not run on hardware
//
#include <hip/hip_runtime.h>

#define VOCAB 1024
#define DP 64
#define N_TOK (32 * 4096)
#define ROWS_PER_WAVE 16
#define WAVES_PER_BLOCK 8
#define ROWS_PER_BLOCK (ROWS_PER_WAVE * WAVES_PER_BLOCK)   // 128
#define N_BLOCKS (N_TOK / ROWS_PER_BLOCK)                  // 1024
#define N_CHUNK (VOCAB / 16)                               // 64

typedef __attribute__((ext_vector_type(8)))  float  v8f;
typedef __attribute__((ext_vector_type(16))) __bf16 v16bf;

union Frag16 {
  uint4          q[2];
  unsigned short s[16];
  v16bf          v;
};

// Truncation-based fp32 -> (bf16_hi, bf16_lo) split: x ~= hi + lo, residual <= 2^-16 * |x|
__device__ __forceinline__ void bf16_split(float x, unsigned short& hi, unsigned short& lo) {
  unsigned u = __float_as_uint(x);
  hi = (unsigned short)(u >> 16);
  float r = x - __uint_as_float(u & 0xffff0000u);
  lo = (unsigned short)(__float_as_uint(r) >> 16);
}

__global__ __launch_bounds__(256, 1)
void vq_main(const float* __restrict__ z, const float* __restrict__ cb,
             float* __restrict__ zq, float* __restrict__ tok_out,
             float* __restrict__ partials) {
  // 320 KB LDS per WGP on CDNA5: stage the full codebook split into bf16 hi/lo planes.
  __shared__ __align__(16) unsigned short s_hi[VOCAB * DP];  // 128 KB
  __shared__ __align__(16) unsigned short s_lo[VOCAB * DP];  // 128 KB
  __shared__ float s_e2[VOCAB];                              //   4 KB
  __shared__ float s_wsum[WAVES_PER_BLOCK];

  const int tid = threadIdx.x;

  // ---- Stage codebook -> LDS (hi/lo split) and e2 table ----
  for (int i = tid; i < VOCAB * DP; i += 256) {
    unsigned short h, l;
    bf16_split(cb[i], h, l);
    s_hi[i] = h;
    s_lo[i] = l;
  }
  for (int k = tid; k < VOCAB; k += 256) {
    const float* row = cb + k * DP;
    float s = 0.f;
    #pragma unroll
    for (int d = 0; d < DP; ++d) s += row[d] * row[d];
    s_e2[k] = s;
  }
  __syncthreads();

  const int wave = tid >> 5;
  const int lane = tid & 31;
  const int n    = lane & 15;   // col-of-tile / code-in-chunk
  const int h    = lane >> 4;   // lane half
  const int row0 = blockIdx.x * ROWS_PER_BLOCK + wave * ROWS_PER_WAVE;

  // ---- Build A fragments for this wave's 16 z-rows (two K=32 halves, hi/lo split) ----
  // A 16x32 frag, contiguous k order used here: lane half h: elems 0..15 -> K=16h+e
  Frag16 Ahi[2], Alo[2];
  const float* zr = z + (size_t)(row0 + n) * DP;   // lane's M-row = lane&15
  #pragma unroll
  for (int f = 0; f < 2; ++f) {
    #pragma unroll
    for (int e = 0; e < 16; ++e) {
      const int d = 32 * f + 16 * h + e;
      bf16_split(zr[d], Ahi[f].s[e], Alo[f].s[e]);
    }
  }

  float best[8];
  int   bidx[8];
  #pragma unroll
  for (int r = 0; r < 8; ++r) { best[r] = 3.4e38f; bidx[r] = 0; }

  // B fragment loader: contiguous k order matching A -> lane n, VGPRs 0..7 hold K = 16h + 0..15
  // frag index: [0]=hi half0, [1]=lo half0, [2]=hi half1, [3]=lo half1
  const int brow = n * DP + 16 * h;
  auto loadB = [&](Frag16* B, int kb) {
    const int cbase = kb * 16 * DP + brow;
    const uint4* ph0 = (const uint4*)&s_hi[cbase];
    const uint4* pl0 = (const uint4*)&s_lo[cbase];
    const uint4* ph1 = (const uint4*)&s_hi[cbase + 32];
    const uint4* pl1 = (const uint4*)&s_lo[cbase + 32];
    B[0].q[0] = ph0[0]; B[0].q[1] = ph0[1];
    B[1].q[0] = pl0[0]; B[1].q[1] = pl0[1];
    B[2].q[0] = ph1[0]; B[2].q[1] = ph1[1];
    B[3].q[0] = pl1[0]; B[3].q[1] = pl1[1];
  };

  // 3 independent accumulator chains (hi*hi, hi*lo, lo*hi) keep 3 WMMAs in flight.
  auto compute = [&](const Frag16* B, int kb) {
    v8f hh = {0.f, 0.f, 0.f, 0.f, 0.f, 0.f, 0.f, 0.f};
    v8f hl = hh, lh = hh;
    hh = __builtin_amdgcn_wmma_f32_16x16x32_bf16(false, Ahi[0].v, false, B[0].v, (short)0, hh, false, false);
    hl = __builtin_amdgcn_wmma_f32_16x16x32_bf16(false, Ahi[0].v, false, B[1].v, (short)0, hl, false, false);
    lh = __builtin_amdgcn_wmma_f32_16x16x32_bf16(false, Alo[0].v, false, B[0].v, (short)0, lh, false, false);
    hh = __builtin_amdgcn_wmma_f32_16x16x32_bf16(false, Ahi[1].v, false, B[2].v, (short)0, hh, false, false);
    hl = __builtin_amdgcn_wmma_f32_16x16x32_bf16(false, Ahi[1].v, false, B[3].v, (short)0, hl, false, false);
    lh = __builtin_amdgcn_wmma_f32_16x16x32_bf16(false, Alo[1].v, false, B[2].v, (short)0, lh, false, false);

    const int   k   = kb * 16 + n;
    const float e2v = s_e2[k];
    #pragma unroll
    for (int r = 0; r < 8; ++r) {
      const float ze   = (hh[r] + hl[r]) + lh[r];
      const float dist = __builtin_fmaf(-2.0f, ze, e2v);   // D tile: m = r + 8h, n = lane&15
      if (dist < best[r]) { best[r] = dist; bidx[r] = k; }
    }
  };

  // ---- Software-pipelined sweep: ping-pong B buffers, prefetch chunk k+1 before computing k ----
  Frag16 B0[4], B1[4];
  loadB(B0, 0);
  for (int kb = 0; kb < N_CHUNK; kb += 2) {
    loadB(B1, kb + 1);          // N_CHUNK even -> always valid
    compute(B0, kb);
    if (kb + 2 < N_CHUNK) loadB(B0, kb + 2);
    compute(B1, kb + 1);
  }

  // ---- Argmin across the 16 lanes of each half (wave32 xor-shuffles), tie -> lowest index ----
  #pragma unroll
  for (int r = 0; r < 8; ++r) {
    float bd = best[r];
    int   bi = bidx[r];
    #pragma unroll
    for (int m = 1; m <= 8; m <<= 1) {
      const float od = __shfl_xor(bd, m, 32);
      const int   oi = __shfl_xor(bi, m, 32);
      if (od < bd || (od == bd && oi < bi)) { bd = od; bi = oi; }
    }
    best[r] = bd;
    bidx[r] = bi;
  }

  // ---- Emit z_q (exact fp32 codebook rows), tokens, and loss partial ----
  float lsum = 0.f;
  #pragma unroll
  for (int m = 0; m < 16; ++m) {
    const int src = (m < 8) ? 0 : 16;              // rows 0-7 live in lanes 0-15, rows 8-15 in 16-31
    const int t   = __shfl(bidx[m & 7], src, 32);
    const int row = row0 + m;
    const float2 cv = ((const float2*)cb)[t * 32 + lane];
    const float2 zv = ((const float2*)z)[(size_t)row * 32 + lane];
    ((float2*)zq)[(size_t)row * 32 + lane] = cv;
    if (lane == 0) tok_out[row] = (float)t;
    const float dx = cv.x - zv.x, dy = cv.y - zv.y;
    lsum += dx * dx + dy * dy;
  }
  #pragma unroll
  for (int m = 1; m <= 16; m <<= 1) lsum += __shfl_xor(lsum, m, 32);
  if (lane == 0) s_wsum[wave] = lsum;
  __syncthreads();
  if (tid == 0) {
    float s = 0.f;
    #pragma unroll
    for (int w = 0; w < WAVES_PER_BLOCK; ++w) s += s_wsum[w];
    partials[blockIdx.x] = s;
  }
}

// Deterministic fixed-order reduction of per-block partials -> commit loss scalar.
__global__ void vq_reduce(const float* __restrict__ partials, int nblk,
                          float* __restrict__ loss_out, float scale) {
  __shared__ float s[256];
  const int t = threadIdx.x;
  float v = 0.f;
  for (int i = t; i < nblk; i += 256) v += partials[i];
  s[t] = v;
  __syncthreads();
  for (int off = 128; off > 0; off >>= 1) {
    if (t < off) s[t] += s[t + off];
    __syncthreads();
  }
  if (t == 0) *loss_out = s[0] * scale;
}

extern "C" void kernel_launch(void* const* d_in, const int* in_sizes, int n_in,
                              void* d_out, int out_size, void* d_ws, size_t ws_size,
                              hipStream_t stream) {
  (void)in_sizes; (void)n_in; (void)out_size; (void)ws_size;
  const float* z  = (const float*)d_in[0];   // [32,4096,64] f32
  const float* cb = (const float*)d_in[1];   // [1024,64]    f32

  float* zq   = (float*)d_out;               // [32,4096,64]
  float* tok  = zq + (size_t)N_TOK * DP;     // [32,4096] (as float)
  float* loss = tok + N_TOK;                 // scalar

  float* partials = (float*)d_ws;            // N_BLOCKS floats

  vq_main<<<N_BLOCKS, 256, 0, stream>>>(z, cb, zq, tok, partials);
  // commit_loss = (1 + BETA) * mean(diff^2),  BETA = 0.25
  vq_reduce<<<1, 256, 0, stream>>>(partials, N_BLOCKS, loss, 1.25f / (float)((size_t)N_TOK * DP));
}
